// Model_23450521436986
// MI455X (gfx1250) — matrix operand order repaired and checked by reading, not compiled, not run on hardware
//
#include <hip/hip_runtime.h>
#include <hip/hip_bf16.h>

// Model dims (compile-time constants from the reference)
#define NN    50000
#define FIN   512
#define HIDC  256
#define GATEC 64
#define EDGES 800000

typedef __attribute__((ext_vector_type(16))) __bf16 v16bf;
typedef __attribute__((ext_vector_type(8)))  float  v8f;

// ---------------------------------------------------------------------------
// Fill kernels (atomic accumulators must be re-zeroed every call).
// ---------------------------------------------------------------------------
__global__ void __launch_bounds__(256) fill4_f32(float4* p, float v, long n4) {
  long i = (long)blockIdx.x * 256 + threadIdx.x;
  if (i < n4) p[i] = make_float4(v, v, v, v);
}
__global__ void __launch_bounds__(256) fill_u32(unsigned* p, unsigned v, long n) {
  long i = (long)blockIdx.x * 256 + threadIdx.x;
  if (i < n) p[i] = v;
}

// ---------------------------------------------------------------------------
// Dense GEMM: C[M,Nc] = A[M,K] @ B[K,Nc] (+bias), fp32 in/out, bf16 WMMA core.
// 256 threads = 8 waves; block covers a 16x128 output tile (one 16x16 per
// wave). Per K=64 step, both operand tiles are cooperatively converted to
// bf16 and staged in LDS:
//   - A (16x64, shared by all 8 waves) in WMMA fragment order
//   - B (64x128) TRANSPOSED to [n][k] so each lane's 16-element fragment is
//     a contiguous 32B run -> ds_load_b128 pair (replaces a serialized
//     16-deep column-stride global load chain per wave)
// Global B staging reads are row-major and coalesced across threads.
// Fragment layouts used here (wave32; A/B k order is contiguous, not the ISA table order):
//   A 16x32 bf16 : row = lane%16 ; elem e -> K = e + 16*(lane>=16)
//   B 32x16 bf16 : col = lane%16 ; elem e -> K = e + 16*(lane>=16)
//   C/D 16x16 f32: col = lane%16 ; elem r -> M = r + 8*(lane>=16)
// Requires K % 64 == 0, M % 16 == 0, Nc % 16 == 0 (true for all call sites).
// ---------------------------------------------------------------------------
__global__ void __launch_bounds__(256)
gemm_bf16_wmma(const float* __restrict__ A, const float* __restrict__ B,
               const float* __restrict__ bias, float* __restrict__ C,
               int Ncols, int K)
{
  const int tid  = threadIdx.x;
  const int lane = tid & 31;
  const int wave = tid >> 5;
  const int mt   = blockIdx.x;                 // M tile
  const int nt   = blockIdx.y * 8 + wave;      // N tile
  const bool active = (nt * 16) < Ncols;       // wave-uniform: EXEC stays full
  const int half = lane >> 4;                  // 0: lanes 0-15, 1: lanes 16-31
  const int lrow = lane & 15;
  const int gc   = nt * 16 + lrow;             // global output column
  const int ln   = wave * 16 + lrow;           // local column in 128-stripe

  __shared__ __align__(32) __bf16 tileA[2][32 * 16];  // fragment order
  __shared__ __align__(32) __bf16 tileB[128][64];     // [local n][k]

  v8f acc = {};

  for (int k0 = 0; k0 < K; k0 += 64) {
    // ---- Stage A (16 x 64) in fragment order: 1024 entries, 4 per thread
#pragma unroll
    for (int s = 0; s < 4; ++s) {
      int idx   = tid * 4 + s;        // 0..1023
      int chunk = idx >> 9;           // 0/1
      int r     = idx & 511;
      int fl    = r >> 4;             // fragment lane 0..31
      int e     = r & 15;             // fragment element 0..15
      int hf    = fl >> 4;
      int row   = fl & 15;
      int k     = k0 + chunk * 32 + e + 16 * hf;
      tileA[chunk][r] = (__bf16)A[(size_t)(mt * 16 + row) * K + k];
    }
    // ---- Stage B (64 x 128) transposed: 4096 even-k pairs, 16 per thread.
    // Consecutive threads touch consecutive columns -> coalesced loads.
#pragma unroll
    for (int s = 0; s < 16; ++s) {
      int idx  = tid + 256 * s;       // 0..4095
      int n    = idx & 127;           // local column
      int k    = (idx >> 7) * 2;      // even k within [0,64)
      int gcol = blockIdx.y * 128 + n;
      if (gcol < Ncols) {
        float f0 = B[(size_t)(k0 + k) * Ncols + gcol];
        float f1 = B[(size_t)(k0 + k + 1) * Ncols + gcol];
        tileB[n][k]     = (__bf16)f0;   // adjacent 2B stores, 4B-aligned pair
        tileB[n][k + 1] = (__bf16)f1;
      }
    }
    __syncthreads();

    if (active) {
#pragma unroll
      for (int chunk = 0; chunk < 2; ++chunk) {
        v16bf a = *reinterpret_cast<const v16bf*>(&tileA[chunk][lane * 16]);
        v16bf b = *reinterpret_cast<const v16bf*>(&tileB[ln][chunk * 32 + 16 * half]);
        // (neg_a, A, neg_b, B, c_mod, C, reuse_a, reuse_b)
        acc = __builtin_amdgcn_wmma_f32_16x16x32_bf16(false, a, false, b,
                                                      (short)0, acc, false, false);
      }
    }
    __syncthreads();
  }

  if (active) {
    const float bv = (bias != nullptr) ? bias[gc] : 0.0f;  // hoisted: one load
    float* Crow = C + (size_t)(mt * 16 + 8 * half) * Ncols + gc;
#pragma unroll
    for (int r = 0; r < 8; ++r)
      Crow[(size_t)r * Ncols] = acc[r] + bv;
  }
}

// ---------------------------------------------------------------------------
// SpMM scatter: out[row[e]] += w[e] * feat[col[e]]  (one block per edge,
// float4 gathered loads, 4 atomics per vector). F4 = F/4, blockDim = 128.
// ---------------------------------------------------------------------------
__global__ void __launch_bounds__(128)
spmm_edges(const int* __restrict__ rows, const int* __restrict__ cols,
           const float* __restrict__ w, const float* __restrict__ feat,
           float* __restrict__ out, int F4)
{
  const int e = blockIdx.x;
  const int r = rows[e], c = cols[e];
  const float we = w[e];
  const float4* src = (const float4*)(feat + (size_t)c * F4 * 4);
  float* dst = out + (size_t)r * F4 * 4;
  __builtin_prefetch(src, 0, 1);   // global_prefetch_b8 on the gathered row
  for (int f = threadIdx.x; f < F4; f += 128) {
    float4 v = src[f];
    atomicAdd(&dst[4 * f + 0], we * v.x);
    atomicAdd(&dst[4 * f + 1], we * v.y);
    atomicAdd(&dst[4 * f + 2], we * v.z);
    atomicAdd(&dst[4 * f + 3], we * v.w);
  }
}

// ---------------------------------------------------------------------------
// Epilogue: v = in + bias; mode 0: prelu(a)  1: relu(prelu(a))  2: relu
// ---------------------------------------------------------------------------
__device__ __forceinline__ float act1(float v, float a, int mode) {
  if (mode == 0 || mode == 1) v = (v >= 0.f) ? v : a * v;
  if (mode == 1 || mode == 2) v = fmaxf(v, 0.f);
  return v;
}

__global__ void __launch_bounds__(256)
bias_act4(const float4* __restrict__ in, const float4* __restrict__ bias,
          const float* __restrict__ alpha, float4* __restrict__ out,
          int F4, long total4, int mode)
{
  long i = (long)blockIdx.x * 256 + threadIdx.x;
  if (i >= total4) return;
  float4 v = in[i];
  float4 bv = bias[(int)(i % F4)];
  const float a = (alpha != nullptr) ? alpha[0] : 0.f;
  v.x = act1(v.x + bv.x, a, mode);
  v.y = act1(v.y + bv.y, a, mode);
  v.z = act1(v.z + bv.z, a, mode);
  v.w = act1(v.w + bv.w, a, mode);
  out[i] = v;
}

// ---------------------------------------------------------------------------
// GAT: alpha_s/alpha_d = h @ att vectors (one wave per node, float4 loads,
// wave32 shuffle reduce)
// ---------------------------------------------------------------------------
__global__ void __launch_bounds__(256)
gat_alpha(const float* __restrict__ h, const float* __restrict__ att_src,
          const float* __restrict__ att_dst, float* __restrict__ as_,
          float* __restrict__ ad_, int n, int F4)
{
  int node = blockIdx.x * 8 + (threadIdx.x >> 5);
  int lane = threadIdx.x & 31;
  if (node >= n) return;
  const float4* row = (const float4*)(h + (size_t)node * F4 * 4);
  const float4* s4  = (const float4*)att_src;
  const float4* d4  = (const float4*)att_dst;
  float s0 = 0.f, s1 = 0.f;
  for (int f = lane; f < F4; f += 32) {
    float4 v = row[f], sa = s4[f], da = d4[f];
    s0 += v.x * sa.x + v.y * sa.y + v.z * sa.z + v.w * sa.w;
    s1 += v.x * da.x + v.y * da.y + v.z * da.z + v.w * da.w;
  }
#pragma unroll
  for (int off = 16; off > 0; off >>= 1) {
    s0 += __shfl_down(s0, off, 32);
    s1 += __shfl_down(s1, off, 32);
  }
  if (lane == 0) { as_[node] = s0; ad_[node] = s1; }
}

// float -> order-preserving unsigned (for atomicMax), and back
__device__ __forceinline__ unsigned f2ord(float f) {
  unsigned u = __float_as_uint(f);
  return (u & 0x80000000u) ? ~u : (u | 0x80000000u);
}
__device__ __forceinline__ float ord2f(unsigned u) {
  return (u & 0x80000000u) ? __uint_as_float(u ^ 0x80000000u) : __uint_as_float(~u);
}

__global__ void __launch_bounds__(256)
gat_edge_logits(const int* __restrict__ rows, const int* __restrict__ cols,
                const float* __restrict__ as_, const float* __restrict__ ad_,
                float* __restrict__ ev, unsigned* __restrict__ mmax)
{
  int e = blockIdx.x * 256 + threadIdx.x;
  if (e >= EDGES) return;
  int r = rows[e];
  float v = as_[cols[e]] + ad_[r];
  v = (v >= 0.f) ? v : 0.2f * v;     // leaky_relu, NEG_SLOPE=0.2
  ev[e] = v;
  atomicMax(&mmax[r], f2ord(v));
}

__global__ void __launch_bounds__(256)
gat_edge_exp(const int* __restrict__ rows, float* __restrict__ ev,
             const unsigned* __restrict__ mmax, float* __restrict__ ssum)
{
  int e = blockIdx.x * 256 + threadIdx.x;
  if (e >= EDGES) return;
  int r = rows[e];
  float p = __expf(ev[e] - ord2f(mmax[r]));
  ev[e] = p;
  atomicAdd(&ssum[r], p);
}

// Attention-weighted scatter, F = HIDC (256) -> 64 float4 per row, blockDim 64
__global__ void __launch_bounds__(64)
gat_spmm(const int* __restrict__ rows, const int* __restrict__ cols,
         const float* __restrict__ p, const float* __restrict__ ssum,
         const float* __restrict__ feat, float* __restrict__ out)
{
  const int e = blockIdx.x;
  const int r = rows[e], c = cols[e];
  const float attn = p[e] / (ssum[r] + 1e-16f);
  const float4* src = (const float4*)(feat + (size_t)c * HIDC);
  float* dst = out + (size_t)r * HIDC;
  __builtin_prefetch(src, 0, 1);
  const int f = threadIdx.x;       // exactly HIDC/4 = 64 vectors
  float4 v = src[f];
  atomicAdd(&dst[4 * f + 0], attn * v.x);
  atomicAdd(&dst[4 * f + 1], attn * v.y);
  atomicAdd(&dst[4 * f + 2], attn * v.z);
  atomicAdd(&dst[4 * f + 3], attn * v.w);
}

// ---------------------------------------------------------------------------
// Gating head: beta = sigmoid([z1 | z2 | deg] @ G3w + G3b)
// ---------------------------------------------------------------------------
__global__ void __launch_bounds__(256)
gate_beta(const float* __restrict__ z1, const float* __restrict__ z2,
          const float* __restrict__ deg, const float* __restrict__ G3w,
          const float* __restrict__ G3b, float* __restrict__ beta)
{
  int i = blockIdx.x * 256 + threadIdx.x;
  if (i >= NN) return;
  float acc = G3b[0] + deg[i] * G3w[2 * GATEC];
  const float4* p1 = (const float4*)(z1 + (size_t)i * GATEC);
  const float4* p2 = (const float4*)(z2 + (size_t)i * GATEC);
  const float4* w1 = (const float4*)G3w;
  const float4* w2 = (const float4*)(G3w + GATEC);
#pragma unroll 4
  for (int g = 0; g < GATEC / 4; ++g) {
    float4 a = p1[g], wa = w1[g];
    float4 b = p2[g], wb = w2[g];
    acc += a.x * wa.x + a.y * wa.y + a.z * wa.z + a.w * wa.w;
    acc += b.x * wb.x + b.y * wb.y + b.z * wb.z + b.w * wb.w;
  }
  beta[i] = 1.f / (1.f + __expf(-acc));
}

// h_gen = hs_gen + beta_gen * hc_gen  (in place; float4, node uniform per vec)
__global__ void __launch_bounds__(256)
combine_gen4(float4* __restrict__ hgen, const float4* __restrict__ hcgen,
             const float* __restrict__ betag, long total4)
{
  long i = (long)blockIdx.x * 256 + threadIdx.x;
  if (i >= total4) return;
  float bg = betag[(int)(i / (HIDC / 4))];
  float4 a = hgen[i], b = hcgen[i];
  a.x += bg * b.x; a.y += bg * b.y; a.z += bg * b.z; a.w += bg * b.w;
  hgen[i] = a;
}

// ---------------------------------------------------------------------------
// One branch: GCN1 -> GCN2 -> GAT
// ---------------------------------------------------------------------------
static void run_branch(const float* x, const int* rows, const int* cols, const float* w,
                       const float* W1, const float* b1, const float* a1,
                       const float* W2, const float* b2, const float* a2,
                       const float* Wg, const float* att_src, const float* att_dst,
                       const float* bg,
                       float* P0, float* P1, float* as_, float* ad_,
                       unsigned* mmax, float* ssum, float* ev,
                       float* h_out, float* hgen_out, hipStream_t stream)
{
  const long n512_4 = (long)NN * 2 * HIDC / 4;
  const long n256_4 = (long)NN * HIDC / 4;

  // GCN1: t1 = x @ W1 [N,512]
  gemm_bf16_wmma<<<dim3(NN / 16, (2 * HIDC) / 128), 256, 0, stream>>>(x, W1, nullptr, P0, 2 * HIDC, FIN);
  fill4_f32<<<(unsigned)((n512_4 + 255) / 256), 256, 0, stream>>>((float4*)P1, 0.f, n512_4);
  spmm_edges<<<EDGES, 128, 0, stream>>>(rows, cols, w, P0, P1, (2 * HIDC) / 4);
  bias_act4<<<(unsigned)((n512_4 + 255) / 256), 256, 0, stream>>>(
      (const float4*)P1, (const float4*)b1, a1, (float4*)P1, (2 * HIDC) / 4, n512_4, /*relu(prelu)*/1);

  // GCN2: t2 = h1 @ W2 [N,256]
  float* t2   = P0;
  float* acc2 = P0 + (size_t)NN * HIDC;
  gemm_bf16_wmma<<<dim3(NN / 16, HIDC / 128), 256, 0, stream>>>(P1, W2, nullptr, t2, HIDC, 2 * HIDC);
  fill4_f32<<<(unsigned)((n256_4 + 255) / 256), 256, 0, stream>>>((float4*)acc2, 0.f, n256_4);
  spmm_edges<<<EDGES, 128, 0, stream>>>(rows, cols, w, t2, acc2, HIDC / 4);
  bias_act4<<<(unsigned)((n256_4 + 255) / 256), 256, 0, stream>>>(
      (const float4*)acc2, (const float4*)b2, a2, (float4*)h_out, HIDC / 4, n256_4, /*prelu*/0);

  // GAT: hg = h_out @ Wg [N,256]
  float* hg   = P1;
  float* gacc = P1 + (size_t)NN * HIDC;
  gemm_bf16_wmma<<<dim3(NN / 16, HIDC / 128), 256, 0, stream>>>(h_out, Wg, nullptr, hg, HIDC, HIDC);
  gat_alpha<<<(NN + 7) / 8, 256, 0, stream>>>(hg, att_src, att_dst, as_, ad_, NN, HIDC / 4);
  fill_u32<<<(NN + 255) / 256, 256, 0, stream>>>(mmax, 0x007FFFFFu, NN);  // encode(-inf)
  fill4_f32<<<(NN / 4 + 255) / 256, 256, 0, stream>>>((float4*)ssum, 0.f, NN / 4);
  gat_edge_logits<<<(EDGES + 255) / 256, 256, 0, stream>>>(rows, cols, as_, ad_, ev, mmax);
  gat_edge_exp<<<(EDGES + 255) / 256, 256, 0, stream>>>(rows, ev, mmax, ssum);
  fill4_f32<<<(unsigned)((n256_4 + 255) / 256), 256, 0, stream>>>((float4*)gacc, 0.f, n256_4);
  gat_spmm<<<EDGES, 64, 0, stream>>>(rows, cols, ev, ssum, hg, gacc);
  bias_act4<<<(unsigned)((n256_4 + 255) / 256), 256, 0, stream>>>(
      (const float4*)gacc, (const float4*)bg, nullptr, (float4*)hgen_out, HIDC / 4, n256_4, /*relu*/2);
}

extern "C" void kernel_launch(void* const* d_in, const int* in_sizes, int n_in,
                              void* d_out, int out_size, void* d_ws, size_t ws_size,
                              hipStream_t stream)
{
  (void)in_sizes; (void)n_in; (void)out_size; (void)ws_size;

  const float* x       = (const float*)d_in[0];
  const int*   ei1     = (const int*)d_in[1];
  const float* ew1     = (const float*)d_in[2];
  const int*   ei2     = (const int*)d_in[3];
  const float* ew2     = (const float*)d_in[4];
  const float* deg     = (const float*)d_in[5];
  const float* W1      = (const float*)d_in[6];
  const float* b1      = (const float*)d_in[7];
  const float* a1      = (const float*)d_in[8];
  const float* W2      = (const float*)d_in[9];
  const float* b2      = (const float*)d_in[10];
  const float* a2      = (const float*)d_in[11];
  const float* Wg      = (const float*)d_in[12];
  const float* att_src = (const float*)d_in[13];
  const float* att_dst = (const float*)d_in[14];
  const float* bg      = (const float*)d_in[15];
  const float* G1w     = (const float*)d_in[16];
  const float* G1b     = (const float*)d_in[17];
  const float* G2w     = (const float*)d_in[18];
  const float* G2b     = (const float*)d_in[19];
  const float* G3w     = (const float*)d_in[20];
  const float* G3b     = (const float*)d_in[21];

  const int* r1 = ei1;          const int* c1 = ei1 + EDGES;
  const int* r2 = ei2;          const int* c2 = ei2 + EDGES;

  // Outputs, concatenated flat in return order: hs, hc, h_gen, beta
  float* out  = (float*)d_out;
  float* hs   = out;
  float* hc   = out + (size_t)NN * HIDC;
  float* hgen = out + 2 * (size_t)NN * HIDC;   // holds hs_gen, combined in place
  float* beta = out + 3 * (size_t)NN * HIDC;

  // Workspace layout
  float*    P0    = (float*)d_ws;                        // N*512
  float*    P1    = P0 + (size_t)NN * 2 * HIDC;          // N*512
  float*    as_   = P1 + (size_t)NN * 2 * HIDC;          // N
  float*    ad_   = as_ + NN;                            // N
  unsigned* mmax  = (unsigned*)(ad_ + NN);               // N
  float*    ssum  = (float*)(mmax + NN);                 // N
  float*    ev    = ssum + NN;                           // E
  float*    z1    = ev + EDGES;                          // N*64
  float*    z2    = z1 + (size_t)NN * GATEC;             // N*64
  float*    betag = z2 + (size_t)NN * GATEC;             // N

  // Branch 1 (adj1): hs, hs_gen(->hgen slot)
  run_branch(x, r1, c1, ew1, W1, b1, a1, W2, b2, a2, Wg, att_src, att_dst, bg,
             P0, P1, as_, ad_, mmax, ssum, ev, hs, hgen, stream);

  // Branch 2 (adj2): hc, hc_gen (parked in P1 low half; free after branch 2)
  float* hcgen = P1;
  run_branch(x, r2, c2, ew2, W1, b1, a1, W2, b2, a2, Wg, att_src, att_dst, bg,
             P0, P1, as_, ad_, mmax, ssum, ev, hc, hcgen, stream);

  // Gating on detached hs/hc
  gemm_bf16_wmma<<<dim3(NN / 16, 1), 256, 0, stream>>>(hs, G1w, G1b, z1, GATEC, HIDC);
  gemm_bf16_wmma<<<dim3(NN / 16, 1), 256, 0, stream>>>(hc, G2w, G2b, z2, GATEC, HIDC);
  gate_beta<<<(NN + 255) / 256, 256, 0, stream>>>(z1, z2, deg, G3w, G3b, beta);

  // Gating on hs_gen/hc_gen, then h_gen = hs_gen + beta_gen * hc_gen
  gemm_bf16_wmma<<<dim3(NN / 16, 1), 256, 0, stream>>>(hgen, G1w, G1b, z1, GATEC, HIDC);
  gemm_bf16_wmma<<<dim3(NN / 16, 1), 256, 0, stream>>>(hcgen, G2w, G2b, z2, GATEC, HIDC);
  gate_beta<<<(NN + 255) / 256, 256, 0, stream>>>(z1, z2, deg, G3w, G3b, betag);

  const long n256_4 = (long)NN * HIDC / 4;
  combine_gen4<<<(unsigned)((n256_4 + 255) / 256), 256, 0, stream>>>(
      (float4*)hgen, (const float4*)hcgen, betag, n256_4);
}
